// MultiModalResNet_38190849196307
// MI455X (gfx1250) — matrix operand order repaired and checked by reading, not compiled, not run on hardware
//
#include <hip/hip_runtime.h>
#include <hip/hip_bf16.h>
#include <math.h>

typedef __bf16 bf16_t;
typedef __attribute__((ext_vector_type(16))) __bf16 v16bf;
typedef __attribute__((ext_vector_type(8)))  float  v8f;
typedef unsigned int u32x4 __attribute__((ext_vector_type(4)));

struct U16x16 { u32x4 lo, hi; };

__device__ __forceinline__ float geluf(float v) {
  return 0.5f * v * (1.0f + erff(v * 0.7071067811865476f));
}

// ============================================================================
// WMMA GEMM: C[b][i,j] = sum_k A[b][i,k] * B[b][j,k]  (+ bias[i])
// A: bf16 [M, lda] row-major, B: bf16 [N, ldb] row-major, K multiple of 32.
// Register-tiled: each wave computes a 32(M) x 64(N) macro-tile = 2x4 WMMA
// tiles, so per K=32 step it loads 2 A frags + 4 B frags (12 x b128) and
// issues 8 v_wmma_f32_16x16x32_bf16 -> 1.5 loads per WMMA.
// Block = 256 threads = 8 waves (2 x 4) covering a 64(M) x 256(N) tile.
// ============================================================================
__global__ __launch_bounds__(256)
void wmma_gemm_kernel(const bf16_t* __restrict__ A, long long aBatch, int lda,
                      const bf16_t* __restrict__ Bm, long long bBatch, int ldb,
                      float* __restrict__ C, long long cBatch, int ldc,
                      const float* __restrict__ bias,
                      int M, int N, int K)
{
  const int b = blockIdx.z;
  const bf16_t* Ab = A + (size_t)b * aBatch;
  const bf16_t* Bb = Bm + (size_t)b * bBatch;
  float* Cb = C + (size_t)b * cBatch;

  const int lane = threadIdx.x & 31;
  const int wave = threadIdx.x >> 5;
  const int i0 = blockIdx.y * 64 + (wave >> 2) * 32;   // wave M origin (2 tiles)
  const int j0 = blockIdx.x * 256 + (wave & 3) * 64;   // wave N origin (4 tiles)
  if (i0 >= M || j0 >= N) return;           // wave-uniform exit, EXEC stays full

  const int half = lane >> 4;               // 0: lanes 0-15, 1: lanes 16-31
  const int l16  = lane & 15;

  // A fragment: lane holds K chunks [base..base+7] and [base+16..base+23], base=8*half
  // B fragment: lane holds K chunks [base..base+7] and [base+16..base+23], base=8*half
  const bf16_t* aptr[2];
  #pragma unroll
  for (int mt = 0; mt < 2; ++mt) {
    int ar = i0 + mt * 16 + l16; if (ar > M - 1) ar = M - 1;   // clamp loads
    aptr[mt] = Ab + (size_t)ar * lda + half * 8;
  }
  const bf16_t* bptr[4];
  #pragma unroll
  for (int nt = 0; nt < 4; ++nt) {
    int br = j0 + nt * 16 + l16; if (br > N - 1) br = N - 1;
    bptr[nt] = Bb + (size_t)br * ldb + half * 8;
  }

  v8f acc[2][4];
  #pragma unroll
  for (int mt = 0; mt < 2; ++mt)
    #pragma unroll
    for (int nt = 0; nt < 4; ++nt)
      acc[mt][nt] = (v8f){};

  for (int k0 = 0; k0 < K; k0 += 32) {
    __builtin_prefetch(aptr[0] + k0 + 64, 0, 1);
    __builtin_prefetch(bptr[0] + k0 + 64, 0, 1);
    v16bf af[2], bf[4];
    #pragma unroll
    for (int mt = 0; mt < 2; ++mt) {
      U16x16 u;
      u.lo = *(const u32x4*)(aptr[mt] + k0);
      u.hi = *(const u32x4*)(aptr[mt] + k0 + 16);
      af[mt] = __builtin_bit_cast(v16bf, u);
    }
    #pragma unroll
    for (int nt = 0; nt < 4; ++nt) {
      U16x16 u;
      u.lo = *(const u32x4*)(bptr[nt] + k0);
      u.hi = *(const u32x4*)(bptr[nt] + k0 + 16);
      bf[nt] = __builtin_bit_cast(v16bf, u);
    }
    #pragma unroll
    for (int mt = 0; mt < 2; ++mt)
      #pragma unroll
      for (int nt = 0; nt < 4; ++nt)
        acc[mt][nt] = __builtin_amdgcn_wmma_f32_16x16x32_bf16(
            false, af[mt], false, bf[nt], (short)0, acc[mt][nt], false, false);
  }

  // D layout: VGPR r -> row tile_i0 + r + 8*half, col tile_j0 + (lane&15)
  #pragma unroll
  for (int mt = 0; mt < 2; ++mt) {
    #pragma unroll
    for (int nt = 0; nt < 4; ++nt) {
      const int j = j0 + nt * 16 + l16;
      #pragma unroll
      for (int r = 0; r < 8; ++r) {
        int i = i0 + mt * 16 + r + half * 8;
        if (i < M && j < N) {
          float v = acc[mt][nt][r];
          if (bias) v += bias[i];
          Cb[(size_t)i * ldc + j] = v;
        }
      }
    }
  }
}

// ============================================================================
// Pack f32 [R, Ccols] -> bf16 [Ccols, Kp] (transposed), optional per-src-row
// instance-norm, per-src-col multiplicative mask, op: 0 none, 1 gelu, 2 relu.
// Zero-fills K padding (r in [R, Kp)).
// ============================================================================
__global__ void pack_T_kernel(const float* __restrict__ src, long long sBatch, int R, int Ccols,
                              bf16_t* __restrict__ dst, long long dBatch, int Kp,
                              const float* __restrict__ mean, const float* __restrict__ rstd,
                              long long stBatch,
                              const float* __restrict__ cmask, long long mBatch, int op)
{
  const int b = blockIdx.z;
  size_t idx = (size_t)blockIdx.x * blockDim.x + threadIdx.x;
  if (idx >= (size_t)Ccols * Kp) return;
  int m = (int)(idx / Kp);
  int r = (int)(idx % Kp);
  float v = 0.0f;
  if (r < R) {
    v = src[(size_t)b * sBatch + (size_t)r * Ccols + m];
    if (mean) v = (v - mean[(size_t)b * stBatch + r]) * rstd[(size_t)b * stBatch + r];
    if (op == 1) v = geluf(v);
    else if (op == 2) v = fmaxf(v, 0.0f);
    if (cmask) v *= cmask[(size_t)b * mBatch + m];
  }
  dst[(size_t)b * dBatch + idx] = (bf16_t)v;
}

// Pack f32 [R, Ccols] -> bf16 [R, Cp] (same orientation), zero-pad cols.
__global__ void pack_N_kernel(const float* __restrict__ src, long long sBatch, int R, int Ccols,
                              bf16_t* __restrict__ dst, long long dBatch, int Cp,
                              const float* __restrict__ mean, const float* __restrict__ rstd,
                              long long stBatch,
                              const float* __restrict__ cmask, long long mBatch, int op)
{
  const int b = blockIdx.z;
  size_t idx = (size_t)blockIdx.x * blockDim.x + threadIdx.x;
  if (idx >= (size_t)R * Cp) return;
  int r = (int)(idx / Cp);
  int c = (int)(idx % Cp);
  float v = 0.0f;
  if (c < Ccols) {
    v = src[(size_t)b * sBatch + (size_t)r * Ccols + c];
    if (mean) v = (v - mean[(size_t)b * stBatch + r]) * rstd[(size_t)b * stBatch + r];
    if (op == 1) v = geluf(v);
    else if (op == 2) v = fmaxf(v, 0.0f);
    if (cmask) v *= cmask[(size_t)b * mBatch + c];
  }
  dst[(size_t)b * dBatch + idx] = (bf16_t)v;
}

// Per-row mean / rsqrt(var+1e-5) over last dim (instance norm stats).
__global__ __launch_bounds__(256)
void rowstats_kernel(const float* __restrict__ src, long long sBatch, int Mcols,
                     float* __restrict__ mean, float* __restrict__ rstd, long long stBatch)
{
  const int b = blockIdx.y;
  const int r = blockIdx.x;
  const float* p = src + (size_t)b * sBatch + (size_t)r * Mcols;
  __shared__ float s1[256], s2[256];
  float a = 0.f, a2 = 0.f;
  for (int m = threadIdx.x; m < Mcols; m += 256) { float v = p[m]; a += v; a2 += v * v; }
  s1[threadIdx.x] = a; s2[threadIdx.x] = a2; __syncthreads();
  for (int s = 128; s > 0; s >>= 1) {
    if (threadIdx.x < s) { s1[threadIdx.x] += s1[threadIdx.x + s]; s2[threadIdx.x] += s2[threadIdx.x + s]; }
    __syncthreads();
  }
  if (threadIdx.x == 0) {
    float mu = s1[0] / Mcols;
    float var = s2[0] / Mcols - mu * mu;
    mean[(size_t)b * stBatch + r] = mu;
    rstd[(size_t)b * stBatch + r] = rsqrtf(var + 1e-5f);
  }
}

// Implicit-GEMM patch extraction (unfold2d) fused with instance norm, -> bf16 [HW, intc*k*k].
__global__ void unfold2d_pack_kernel(const float* __restrict__ x1, long long sBatch,
                                     int intc, int Hh, int Ww,
                                     const float* __restrict__ mean, const float* __restrict__ rstd,
                                     long long stBatch,
                                     bf16_t* __restrict__ dst, long long dBatch, int kk)
{
  const int b = blockIdx.z;
  const int Kd = intc * kk * kk;
  size_t idx = (size_t)blockIdx.x * blockDim.x + threadIdx.x;
  if (idx >= (size_t)Hh * Ww * Kd) return;
  int m  = (int)(idx / Kd);
  int t  = (int)(idx % Kd);
  int ci = t / (kk * kk);
  int tt = t % (kk * kk);
  int y = m / Ww + tt / kk - kk / 2;
  int x = m % Ww + tt % kk - kk / 2;
  float v = 0.0f;
  if (y >= 0 && y < Hh && x >= 0 && x < Ww) {
    v = x1[(size_t)b * sBatch + ((size_t)ci * Hh + y) * Ww + x];
    v = (v - mean[(size_t)b * stBatch + ci]) * rstd[(size_t)b * stBatch + ci];
  }
  dst[(size_t)b * dBatch + idx] = (bf16_t)v;
}

// unfold1d with replicate padding -> bf16 [NL, Kp], zero-fill K padding.
__global__ void unfold1d_pack_kernel(const float* __restrict__ l1, long long sBatch,
                                     int intc, int Nl,
                                     bf16_t* __restrict__ dst, long long dBatch, int kk, int Kp)
{
  const int b = blockIdx.z;
  size_t idx = (size_t)blockIdx.x * blockDim.x + threadIdx.x;
  if (idx >= (size_t)Nl * Kp) return;
  int n = (int)(idx / Kp);
  int t = (int)(idx % Kp);
  float v = 0.0f;
  int K = intc * kk;
  if (t < K) {
    int ci = t / kk;
    int i  = t % kk;
    int sn = n + i - kk / 2;
    sn = sn < 0 ? 0 : (sn > Nl - 1 ? Nl - 1 : sn);
    v = l1[(size_t)b * sBatch + (size_t)ci * Nl + sn];
  }
  dst[(size_t)b * dBatch + idx] = (bf16_t)v;
}

// softmax of the two 3-element mixing weight vectors.
__global__ void softmax3_kernel(const float* __restrict__ vw, const float* __restrict__ lw,
                                float* __restrict__ out)
{
  if (threadIdx.x == 0 && blockIdx.x == 0) {
    for (int s = 0; s < 2; ++s) {
      const float* p = s ? lw : vw;
      float m = fmaxf(p[0], fmaxf(p[1], p[2]));
      float e0 = expf(p[0] - m), e1 = expf(p[1] - m), e2 = expf(p[2] - m);
      float sum = e0 + e1 + e2;
      out[s * 3 + 0] = e0 / sum; out[s * 3 + 1] = e1 / sum; out[s * 3 + 2] = e2 / sum;
    }
  }
}

// Row softmax with optional additive key mask; dst = (accum? dst:0) + wvec[widx]*softmax.
__global__ __launch_bounds__(256)
void softmax_accum_kernel(const float* __restrict__ src, long long sBatch, int Ccols,
                          float scl, const float* __restrict__ cmask, long long mBatch,
                          const float* __restrict__ wvec, int widx, int accum,
                          float* __restrict__ dst, long long dBatch)
{
  const int b = blockIdx.y;
  const int r = blockIdx.x;
  const float* p = src + (size_t)b * sBatch + (size_t)r * Ccols;
  float* q = dst + (size_t)b * dBatch + (size_t)r * Ccols;
  __shared__ float sh[256];
  float mx = -3.0e38f;
  for (int c = threadIdx.x; c < Ccols; c += 256) {
    float v = p[c] * scl;
    if (cmask) v += 10000.0f * cmask[(size_t)b * mBatch + c] - 10000.0f;
    mx = fmaxf(mx, v);
  }
  sh[threadIdx.x] = mx; __syncthreads();
  for (int s = 128; s > 0; s >>= 1) {
    if (threadIdx.x < s) sh[threadIdx.x] = fmaxf(sh[threadIdx.x], sh[threadIdx.x + s]);
    __syncthreads();
  }
  mx = sh[0]; __syncthreads();
  float sum = 0.0f;
  for (int c = threadIdx.x; c < Ccols; c += 256) {
    float v = p[c] * scl;
    if (cmask) v += 10000.0f * cmask[(size_t)b * mBatch + c] - 10000.0f;
    sum += expf(v - mx);
  }
  sh[threadIdx.x] = sum; __syncthreads();
  for (int s = 128; s > 0; s >>= 1) {
    if (threadIdx.x < s) sh[threadIdx.x] += sh[threadIdx.x + s];
    __syncthreads();
  }
  float inv = wvec[widx] / sh[0];
  for (int c = threadIdx.x; c < Ccols; c += 256) {
    float v = p[c] * scl;
    if (cmask) v += 10000.0f * cmask[(size_t)b * mBatch + c] - 10000.0f;
    float e = expf(v - mx) * inv;
    q[c] = accum ? q[c] + e : e;
  }
}

// PWAM input: dst[m,c] = gelu(visq[c,m]) * inorm(outv2)[c,m]  (bf16, transposed)
__global__ void mmin_kernel(const float* __restrict__ visq, const float* __restrict__ outv2,
                            long long sBatch,
                            const float* __restrict__ mean, const float* __restrict__ rstd,
                            long long stBatch,
                            bf16_t* __restrict__ dst, long long dBatch, int Cc, int Mc)
{
  const int b = blockIdx.z;
  size_t idx = (size_t)blockIdx.x * blockDim.x + threadIdx.x;
  if (idx >= (size_t)Cc * Mc) return;
  int c = (int)(idx / Mc);
  int m = (int)(idx % Mc);
  float v = geluf(visq[(size_t)b * sBatch + idx]);
  float o = (outv2[(size_t)b * sBatch + idx] - mean[(size_t)b * stBatch + c]) *
            rstd[(size_t)b * stBatch + c];
  dst[(size_t)b * dBatch + (size_t)m * Cc + c] = (bf16_t)(v * o);
}

__global__ void final_x_kernel(const float* __restrict__ x, const float* __restrict__ r2,
                               const float* __restrict__ mm, float* __restrict__ out, size_t total)
{
  size_t i = (size_t)blockIdx.x * blockDim.x + threadIdx.x;
  if (i < total) out[i] = x[i] + tanhf(r2[i]) * geluf(mm[i]);
}

__global__ void final_l_kernel(const float* __restrict__ l, const float* __restrict__ wl,
                               float* __restrict__ out, size_t total)
{
  size_t i = (size_t)blockIdx.x * blockDim.x + threadIdx.x;
  if (i < total) out[i] = l[i] + geluf(wl[i]);
}

// ============================================================================
extern "C" void kernel_launch(void* const* d_in, const int* in_sizes, int n_in,
                              void* d_out, int out_size, void* d_ws, size_t ws_size,
                              hipStream_t stream)
{
  (void)in_sizes; (void)n_in; (void)out_size; (void)ws_size;
  const int B = 4, H = 48, W = 48, HW = 2304, NL = 20;
  const float scale = 1.0f / sqrtf(512.0f);

  const float* x     = (const float*)d_in[0];
  const float* l     = (const float*)d_in[1];
  const float* lmask = (const float*)d_in[2];      // [B, NL, 1]
  auto P = [&](int i) { return (const float*)d_in[i]; };

  size_t off = 0;
  auto alloc = [&](size_t bytes) -> void* {
    off = (off + 255) & ~(size_t)255;
    void* p = (char*)d_ws + off;
    off += bytes;
    return p;
  };
  auto allocF = [&](size_t n) { return (float*)alloc(n * 4); };
  auto allocB = [&](size_t n) { return (bf16_t*)alloc(n * 2); };

  auto gemm = [&](const bf16_t* A, long long aB, int lda, const bf16_t* Bm, long long bB, int ldb,
                  float* C, long long cB, int ldc, const float* bias, int M, int N, int K) {
    dim3 g((unsigned)((N + 255) / 256), (unsigned)((M + 63) / 64), B);
    wmma_gemm_kernel<<<g, 256, 0, stream>>>(A, aB, lda, Bm, bB, ldb, C, cB, ldc, bias, M, N, K);
  };
  auto packT = [&](const float* src, long long sB, int R, int C, bf16_t* dst, int Kp,
                   const float* mean, const float* rstd, long long stB,
                   const float* cm, long long mB, int op) {
    size_t tot = (size_t)C * Kp;
    dim3 g((unsigned)((tot + 255) / 256), 1, B);
    pack_T_kernel<<<g, 256, 0, stream>>>(src, sB, R, C, dst, (long long)tot, Kp,
                                         mean, rstd, stB, cm, mB, op);
  };
  auto packN = [&](const float* src, long long sB, int R, int C, bf16_t* dst, int Cp,
                   const float* mean, const float* rstd, long long stB,
                   const float* cm, long long mB, int op, int nb) {
    size_t tot = (size_t)R * Cp;
    dim3 g((unsigned)((tot + 255) / 256), 1, nb);
    pack_N_kernel<<<g, 256, 0, stream>>>(src, sB, R, C, dst, (long long)tot, Cp,
                                         mean, rstd, stB, cm, mB, op);
  };
  auto packWeight = [&](int idx, int Cout, int Cin) -> bf16_t* {
    int Cp = (Cin + 31) & ~31;
    bf16_t* wp = allocB((size_t)Cout * Cp);
    packN(P(idx), 0, Cout, Cin, wp, Cp, nullptr, nullptr, 0, nullptr, 0, 0, 1);
    return wp;
  };

  // ---------------- persistent: packed weights & activations ----------------
  const int vK[3] = {1, 3, 5}, vIC[3] = {512, 256, 128}, vIdx[3] = {3, 9, 15};
  const int lK[3] = {1, 2, 3}, lIC[3] = {512, 256, 170}, lIdx[3] = {21, 27, 33};

  bf16_t *WvFq[3], *WvFq2[3], *WvFk[3], *WlFq[3], *WlFq2[3], *WlFk[3];
  for (int i = 0; i < 3; ++i) {
    WvFq[i]  = packWeight(vIdx[i] + 0, vIC[i], 512);
    WvFq2[i] = packWeight(vIdx[i] + 2, 512, vIC[i] * vK[i] * vK[i]);
    WvFk[i]  = packWeight(vIdx[i] + 4, 512, 768);
    WlFq[i]  = packWeight(lIdx[i] + 0, lIC[i], 768);
    WlFq2[i] = packWeight(lIdx[i] + 2, 768, lIC[i] * lK[i]);
    WlFk[i]  = packWeight(lIdx[i] + 4, 768, 512);
  }
  bf16_t* WfvP  = packWeight(39, 512, 768);
  bf16_t* WfvvP = packWeight(41, 768, 512);
  bf16_t* WWvP  = packWeight(43, 512, 512);
  bf16_t* WW2P  = packWeight(45, 768, 768);
  bf16_t* WvpP  = packWeight(49, 512, 512);
  bf16_t* WpmP  = packWeight(51, 512, 512);
  bf16_t* Wg1P  = packWeight(53, 512, 512);
  bf16_t* Wg2P  = packWeight(54, 512, 512);
  bf16_t* WWlP  = packWeight(55, 768, 768);

  bf16_t* Xt = allocB((size_t)B * HW * 512);          // [HW,512] per batch
  packT(x, (long long)512 * HW, 512, HW, Xt, 512, nullptr, nullptr, 0, nullptr, 0, 0);
  bf16_t* Lt = allocB((size_t)B * NL * 768);          // [NL,768] per batch
  packT(l, (long long)768 * NL, 768, NL, Lt, 768, nullptr, nullptr, 0, nullptr, 0, 0);

  float* ws6 = allocF(8);
  softmax3_kernel<<<1, 32, 0, stream>>>(P(47), P(48), ws6);

  float* simv   = allocF((size_t)B * HW * NL);
  float* siml   = allocF((size_t)B * NL * HW);
  float* outv2  = allocF((size_t)B * 512 * HW);
  float* ovMean = allocF((size_t)B * 512);
  float* ovRstd = allocF((size_t)B * 512);

  size_t mark1 = off;

  // ---------------- visual similarity branches (k = 1,3,5) ----------------
  for (int vi = 0; vi < 3; ++vi) {
    off = mark1;
    const int kk = vK[vi], ic = vIC[vi], Kd = ic * kk * kk;
    float* x1 = allocF((size_t)B * ic * HW);
    gemm(WvFq[vi], 0, 512, Xt, (long long)HW * 512, 512,
         x1, (long long)ic * HW, HW, P(vIdx[vi] + 1), ic, HW, 512);
    float* m1 = allocF((size_t)B * ic); float* r1s = allocF((size_t)B * ic);
    rowstats_kernel<<<dim3(ic, B), 256, 0, stream>>>(x1, (long long)ic * HW, HW, m1, r1s, ic);
    bf16_t* pat = allocB((size_t)B * HW * Kd);
    { size_t tot = (size_t)HW * Kd; dim3 g((unsigned)((tot + 255) / 256), 1, B);
      unfold2d_pack_kernel<<<g, 256, 0, stream>>>(x1, (long long)ic * HW, ic, H, W,
                                                  m1, r1s, ic, pat, (long long)tot, kk); }
    float* q = allocF((size_t)B * 512 * HW);
    gemm(WvFq2[vi], 0, Kd, pat, (long long)HW * Kd, Kd,
         q, (long long)512 * HW, HW, P(vIdx[vi] + 3), 512, HW, Kd);
    float* mq = allocF((size_t)B * 512); float* rq = allocF((size_t)B * 512);
    rowstats_kernel<<<dim3(512, B), 256, 0, stream>>>(q, (long long)512 * HW, HW, mq, rq, 512);
    bf16_t* Qt = allocB((size_t)B * HW * 512);
    packT(q, (long long)512 * HW, 512, HW, Qt, 512, mq, rq, 512, nullptr, 0, 0);
    float* kkv = allocF((size_t)B * 512 * NL);
    gemm(WvFk[vi], 0, 768, Lt, (long long)NL * 768, 768,
         kkv, (long long)512 * NL, NL, P(vIdx[vi] + 5), 512, NL, 768);
    bf16_t* KKt = allocB((size_t)B * NL * 512);
    packT(kkv, (long long)512 * NL, 512, NL, KKt, 512, nullptr, nullptr, 0, lmask, NL, 0);
    float* s = allocF((size_t)B * HW * NL);
    gemm(Qt, (long long)HW * 512, 512, KKt, (long long)NL * 512, 512,
         s, (long long)HW * NL, NL, nullptr, HW, NL, 512);
    softmax_accum_kernel<<<dim3(HW, B), 256, 0, stream>>>(s, (long long)HW * NL, NL, scale,
        lmask, NL, ws6, vi, vi > 0 ? 1 : 0, simv, (long long)HW * NL);
  }

  // ---------------- language similarity branches (k = 1,2,3) ----------------
  for (int li = 0; li < 3; ++li) {
    off = mark1;
    const int kk = lK[li], ic = lIC[li];
    const int K1 = ic * kk, Kp1 = (K1 + 31) & ~31;
    float* l1 = allocF((size_t)B * ic * NL);
    gemm(WlFq[li], 0, 768, Lt, (long long)NL * 768, 768,
         l1, (long long)ic * NL, NL, P(lIdx[li] + 1), ic, NL, 768);
    bf16_t* pat1 = allocB((size_t)B * NL * Kp1);
    { size_t tot = (size_t)NL * Kp1; dim3 g((unsigned)((tot + 255) / 256), 1, B);
      unfold1d_pack_kernel<<<g, 256, 0, stream>>>(l1, (long long)ic * NL, ic, NL,
                                                  pat1, (long long)tot, kk, Kp1); }
    float* ql = allocF((size_t)B * 768 * NL);
    gemm(WlFq2[li], 0, Kp1, pat1, (long long)NL * Kp1, Kp1,
         ql, (long long)768 * NL, NL, P(lIdx[li] + 3), 768, NL, Kp1);
    bf16_t* Qlt = allocB((size_t)B * NL * 768);
    packT(ql, (long long)768 * NL, 768, NL, Qlt, 768, nullptr, nullptr, 0, lmask, NL, 0);
    float* kkl = allocF((size_t)B * 768 * HW);
    gemm(WlFk[li], 0, 512, Xt, (long long)HW * 512, 512,
         kkl, (long long)768 * HW, HW, P(lIdx[li] + 5), 768, HW, 512);
    float* mk = allocF((size_t)B * 768); float* rk = allocF((size_t)B * 768);
    rowstats_kernel<<<dim3(768, B), 256, 0, stream>>>(kkl, (long long)768 * HW, HW, mk, rk, 768);
    bf16_t* KKlt = allocB((size_t)B * HW * 768);
    packT(kkl, (long long)768 * HW, 768, HW, KKlt, 768, mk, rk, 768, nullptr, 0, 0);
    float* sl = allocF((size_t)B * NL * HW);
    gemm(Qlt, (long long)NL * 768, 768, KKlt, (long long)HW * 768, 768,
         sl, (long long)NL * HW, HW, nullptr, NL, HW, 768);
    // lan query-row mask is an additive per-row constant -> softmax-invariant, omitted
    softmax_accum_kernel<<<dim3(NL, B), 256, 0, stream>>>(sl, (long long)NL * HW, HW, scale,
        nullptr, 0, ws6 + 3, li, li > 0 ? 1 : 0, siml, (long long)NL * HW);
  }

  // ---------------- visual attention output: out_v = inorm(Wv(simv @ value^T)) ----------------
  off = mark1;
  float* value = allocF((size_t)B * 512 * NL);
  gemm(WfvP, 0, 768, Lt, (long long)NL * 768, 768,
       value, (long long)512 * NL, NL, P(40), 512, NL, 768);
  bf16_t* valueP = allocB((size_t)B * 512 * 32);
  packN(value, (long long)512 * NL, 512, NL, valueP, 32, nullptr, nullptr, 0, lmask, NL, 0, B);
  bf16_t* SimvP = allocB((size_t)B * HW * 32);
  packN(simv, (long long)HW * NL, HW, NL, SimvP, 32, nullptr, nullptr, 0, nullptr, 0, 0, B);
  float* outv = allocF((size_t)B * 512 * HW);
  gemm(valueP, (long long)512 * 32, 32, SimvP, (long long)HW * 32, 32,
       outv, (long long)512 * HW, HW, nullptr, 512, HW, 32);
  bf16_t* OutVt = allocB((size_t)B * HW * 512);
  packT(outv, (long long)512 * HW, 512, HW, OutVt, 512, nullptr, nullptr, 0, nullptr, 0, 0);
  gemm(WWvP, 0, 512, OutVt, (long long)HW * 512, 512,
       outv2, (long long)512 * HW, HW, P(44), 512, HW, 512);
  rowstats_kernel<<<dim3(512, B), 256, 0, stream>>>(outv2, (long long)512 * HW, HW,
                                                    ovMean, ovRstd, 512);

  // ---------------- language attention output + l_out ----------------
  float* vv = allocF((size_t)B * 768 * HW);
  gemm(WfvvP, 0, 512, Xt, (long long)HW * 512, 512,
       vv, (long long)768 * HW, HW, P(42), 768, HW, 512);
  float* mvv = allocF((size_t)B * 768); float* rvv = allocF((size_t)B * 768);
  rowstats_kernel<<<dim3(768, B), 256, 0, stream>>>(vv, (long long)768 * HW, HW, mvv, rvv, 768);
  bf16_t* VvP = allocB((size_t)B * 768 * HW);
  packN(vv, (long long)768 * HW, 768, HW, VvP, HW, mvv, rvv, 768, nullptr, 0, 0, B);
  bf16_t* SimlP = allocB((size_t)B * NL * HW);
  packN(siml, (long long)NL * HW, NL, HW, SimlP, HW, nullptr, nullptr, 0, nullptr, 0, 0, B);
  float* outl = allocF((size_t)B * 768 * NL);
  gemm(VvP, (long long)768 * HW, HW, SimlP, (long long)NL * HW, HW,
       outl, (long long)768 * NL, NL, nullptr, 768, NL, HW);
  bf16_t* OutLt = allocB((size_t)B * NL * 768);
  packT(outl, (long long)768 * NL, 768, NL, OutLt, 768, nullptr, nullptr, 0, nullptr, 0, 0);
  float* outl2 = allocF((size_t)B * 768 * NL);
  gemm(WW2P, 0, 768, OutLt, (long long)NL * 768, 768,
       outl2, (long long)768 * NL, NL, P(46), 768, NL, 768);
  bf16_t* OutL2t = allocB((size_t)B * NL * 768);
  packT(outl2, (long long)768 * NL, 768, NL, OutL2t, 768, nullptr, nullptr, 0, nullptr, 0, 0);
  float* wlout = allocF((size_t)B * 768 * NL);
  gemm(WWlP, 0, 768, OutL2t, (long long)NL * 768, 768,
       wlout, (long long)768 * NL, NL, P(56), 768, NL, 768);
  { size_t tot = (size_t)B * 768 * NL; dim3 g((unsigned)((tot + 255) / 256));
    final_l_kernel<<<g, 256, 0, stream>>>(l, wlout, (float*)d_out + (size_t)B * 512 * HW, tot); }

  // ---------------- PWAM + residual gate -> x_out ----------------
  off = mark1;
  float* visq = allocF((size_t)B * 512 * HW);
  gemm(WvpP, 0, 512, Xt, (long long)HW * 512, 512,
       visq, (long long)512 * HW, HW, P(50), 512, HW, 512);
  bf16_t* MmT = allocB((size_t)B * HW * 512);
  { size_t tot = (size_t)512 * HW; dim3 g((unsigned)((tot + 255) / 256), 1, B);
    mmin_kernel<<<g, 256, 0, stream>>>(visq, outv2, (long long)512 * HW,
                                       ovMean, ovRstd, 512, MmT, (long long)HW * 512, 512, HW); }
  float* mm = allocF((size_t)B * 512 * HW);
  gemm(WpmP, 0, 512, MmT, (long long)HW * 512, 512,
       mm, (long long)512 * HW, HW, P(52), 512, HW, 512);
  bf16_t* XrT = allocB((size_t)B * HW * 512);
  packT(mm, (long long)512 * HW, 512, HW, XrT, 512, nullptr, nullptr, 0, nullptr, 0, 1);
  float* r1 = allocF((size_t)B * 512 * HW);
  gemm(Wg1P, 0, 512, XrT, (long long)HW * 512, 512,
       r1, (long long)512 * HW, HW, nullptr, 512, HW, 512);
  bf16_t* R1t = allocB((size_t)B * HW * 512);
  packT(r1, (long long)512 * HW, 512, HW, R1t, 512, nullptr, nullptr, 0, nullptr, 0, 2);
  float* r2 = allocF((size_t)B * 512 * HW);
  gemm(Wg2P, 0, 512, R1t, (long long)HW * 512, 512,
       r2, (long long)512 * HW, HW, nullptr, 512, HW, 512);
  { size_t tot = (size_t)B * 512 * HW; dim3 g((unsigned)((tot + 255) / 256));
    final_x_kernel<<<g, 256, 0, stream>>>(x, r2, mm, (float*)d_out, tot); }
}
